// GraphNetRoberta_90142773608885
// MI455X (gfx1250) — matrix operand order repaired and checked by reading, not compiled, not run on hardware
//
#include <hip/hip_runtime.h>
#include <stdint.h>

// ---------------------------------------------------------------------------
// Types for CDNA5 WMMA (wave32): v16bf A/B fragments, v8f accumulator
// ---------------------------------------------------------------------------
typedef __attribute__((ext_vector_type(16))) __bf16    v16bf;
typedef __attribute__((ext_vector_type(8)))  float     v8f;
typedef __attribute__((ext_vector_type(4)))  uint32_t  v4u;

union FragBF { v16bf v; v4u q[2]; uint32_t u[8]; };
union FragF  { v8f   v; float f[8]; };

// round-to-nearest-even f32 -> bf16 (bit-level)
static __device__ __forceinline__ uint16_t f2bf(float f) {
  union { float f; uint32_t u; } v; v.f = f;
  uint32_t r = v.u + 0x7FFFu + ((v.u >> 16) & 1u);
  if ((v.u & 0x7F800000u) == 0x7F800000u) r = v.u;   // inf/nan passthrough
  return (uint16_t)(r >> 16);
}

// ---------------------------------------------------------------------------
// Row-padded f32 -> bf16 convert: dst[r][0..ldd) = src[r][0..lds) zero-padded
// ---------------------------------------------------------------------------
__global__ __launch_bounds__(256)
void k_pad_bf(const float* __restrict__ src, uint16_t* __restrict__ dst,
              int rows, int lds, int ldd) {
  long long idx = (long long)blockIdx.x * blockDim.x + threadIdx.x;
  if (idx >= (long long)rows * ldd) return;
  int r = (int)(idx / ldd), c = (int)(idx % ldd);
  float v = (c < lds) ? src[(size_t)r * lds + c] : 0.0f;
  dst[idx] = f2bf(v);
}

// ---------------------------------------------------------------------------
// Weight transpose + convert: W[K x N] f32 -> Bt[Npad x Kpad] bf16, zero pad
// ---------------------------------------------------------------------------
__global__ __launch_bounds__(256)
void k_wt_bf(const float* __restrict__ W, uint16_t* __restrict__ Bt,
             int K, int N, int Kpad, int Npad) {
  long long idx = (long long)blockIdx.x * blockDim.x + threadIdx.x;
  if (idx >= (long long)Npad * Kpad) return;
  int n = (int)(idx / Kpad), k = (int)(idx % Kpad);
  float v = (n < N && k < K) ? W[(size_t)k * N + n] : 0.0f;
  Bt[idx] = f2bf(v);
}

// ---------------------------------------------------------------------------
// deg[col[e]] += w[e]  (in-degree over targets; self-loop +1 folded in k_dinv)
// ---------------------------------------------------------------------------
__global__ __launch_bounds__(256)
void k_deg(const long long* __restrict__ cols, const float* __restrict__ ew,
           float* __restrict__ deg, long long E) {
  long long e = (long long)blockIdx.x * blockDim.x + threadIdx.x;
  if (e < E) atomicAdd(&deg[cols[e]], ew[e]);
}

__global__ __launch_bounds__(256)
void k_dinv(const float* __restrict__ deg, float* __restrict__ dinv, int n) {
  int i = blockIdx.x * blockDim.x + threadIdx.x;
  if (i < n) {
    float d = deg[i] + 1.0f;                 // +1: self-loop weight
    dinv[i] = (d > 0.0f) ? rsqrtf(d) : 0.0f;
  }
}

// ---------------------------------------------------------------------------
// bf16 WMMA GEMM, no LDS:
//   C[M x Ncols] = A[M x Kpad](bf16, row-major, K zero-padded)
//                * Bt[Npad x Kpad](bf16, weights pre-transposed+padded) (+bias)
// 256 threads = 8 waves; block tile 128 rows x 64 cols; wave w owns rows
// [16w,16w+16), 4 accumulator tiles across 64 cols.
// A fragment  (16-bit A 16x32): lane holds row (lane&15); K chunk
//   [k0+16h, +16) contiguous  (h = lane>>4)  -> two b128 loads.
// B fragment  (16-bit B 32x16): lane holds col (lane&15); K chunk
//   [k0+16h, +16) contiguous in Bt  -> two b128 loads.
// All strides are 16B multiples (Kpad % 32 == 0), so loads are b128-aligned.
// Rows >= M are clamped (their results are computed but never stored).
// ---------------------------------------------------------------------------
#define BM 128
#define BN 64

__global__ __launch_bounds__(256)
void k_gemm_bf16_wmma(const uint16_t* __restrict__ A,
                      const uint16_t* __restrict__ Bt,
                      const float* __restrict__ bias,
                      float* __restrict__ Cf, uint16_t* __restrict__ Cbf,
                      int ldc, int M, int Ncols, int Kpad)
{
  const int wave = threadIdx.x >> 5;
  const int lane = threadIdx.x & 31;
  const int half = lane >> 4;
  const int ml   = lane & 15;

  const int rowBase = blockIdx.y * BM;
  const int colBase = blockIdx.x * BN;

  int row  = rowBase + wave * 16 + ml;
  int rowc = row < M ? row : (M - 1);                 // clamp for safe loads
  const uint16_t* Arow = A + (size_t)rowc * Kpad;

  const uint16_t* Brow0 = Bt + (size_t)(colBase + 0 * 16 + ml) * Kpad;
  const uint16_t* Brow1 = Bt + (size_t)(colBase + 1 * 16 + ml) * Kpad;
  const uint16_t* Brow2 = Bt + (size_t)(colBase + 2 * 16 + ml) * Kpad;
  const uint16_t* Brow3 = Bt + (size_t)(colBase + 3 * 16 + ml) * Kpad;

  const v8f zero = {0, 0, 0, 0, 0, 0, 0, 0};
  v8f acc0 = zero, acc1 = zero, acc2 = zero, acc3 = zero;

  for (int k0 = 0; k0 < Kpad; k0 += 32) {
    FragBF a;
    a.q[0] = *(const v4u*)(Arow + k0 + half * 16);
    a.q[1] = *(const v4u*)(Arow + k0 + half * 16 + 8);

    FragBF b0, b1, b2, b3;
    b0.q[0] = *(const v4u*)(Brow0 + k0 + half * 16);
    b0.q[1] = *(const v4u*)(Brow0 + k0 + half * 16 + 8);
    b1.q[0] = *(const v4u*)(Brow1 + k0 + half * 16);
    b1.q[1] = *(const v4u*)(Brow1 + k0 + half * 16 + 8);
    b2.q[0] = *(const v4u*)(Brow2 + k0 + half * 16);
    b2.q[1] = *(const v4u*)(Brow2 + k0 + half * 16 + 8);
    b3.q[0] = *(const v4u*)(Brow3 + k0 + half * 16);
    b3.q[1] = *(const v4u*)(Brow3 + k0 + half * 16 + 8);

    acc0 = __builtin_amdgcn_wmma_f32_16x16x32_bf16(false, a.v, false, b0.v,
                                                   (short)0, acc0, false, false);
    acc1 = __builtin_amdgcn_wmma_f32_16x16x32_bf16(false, a.v, false, b1.v,
                                                   (short)0, acc1, false, false);
    acc2 = __builtin_amdgcn_wmma_f32_16x16x32_bf16(false, a.v, false, b2.v,
                                                   (short)0, acc2, false, false);
    acc3 = __builtin_amdgcn_wmma_f32_16x16x32_bf16(false, a.v, false, b3.v,
                                                   (short)0, acc3, false, false);
  }

  // C/D layout: lane -> col = lane&15; VGPR j -> row j (+8 for lanes 16-31)
  v8f accs[4] = {acc0, acc1, acc2, acc3};
#pragma unroll
  for (int t = 0; t < 4; ++t) {
    int gc = colBase + t * 16 + ml;
    if (gc >= Ncols) continue;
    float bv = bias ? bias[gc] : 0.0f;
    FragF c; c.v = accs[t];
#pragma unroll
    for (int j = 0; j < 8; ++j) {
      int gr = rowBase + wave * 16 + half * 8 + j;
      if (gr < M) {
        float val = c.f[j] + bv;
        if (Cf)  Cf [(size_t)gr * ldc + gc] = val;
        if (Cbf) Cbf[(size_t)gr * ldc + gc] = f2bf(val);
      }
    }
  }
}

// ---------------------------------------------------------------------------
// out[i][f] = bias[f] + dinv[i]^2 * xw[i][f]   (self-loop + bias; non-atomic
// initializer that also replaces the memset before edge atomics)
// ---------------------------------------------------------------------------
__global__ __launch_bounds__(256)
void k_selfinit(const float* __restrict__ xw, const float* __restrict__ dinv,
                const float* __restrict__ bias, float* __restrict__ out,
                int n, int F) {
  long long idx = (long long)blockIdx.x * blockDim.x + threadIdx.x;
  if (idx < (long long)n * F) {
    int i = (int)(idx / F), f = (int)(idx % F);
    float di = dinv[i];
    out[idx] = (bias ? bias[f] : 0.0f) + di * di * xw[idx];
  }
}

// ---------------------------------------------------------------------------
// Edge scatter: one wave32 per edge; out[c] += dinv[r]*w*dinv[c] * xw[r].
// xw + out fit the 192MB L2, so atomics mostly resolve in L2.
// ---------------------------------------------------------------------------
__global__ __launch_bounds__(256)
void k_scatter_edges(const long long* __restrict__ rows,
                     const long long* __restrict__ cols,
                     const float* __restrict__ ew,
                     const float* __restrict__ dinv,
                     const float* __restrict__ xw,
                     float* __restrict__ out,
                     long long E, int F)
{
  long long e = (long long)blockIdx.x * 8 + (threadIdx.x >> 5);
  if (e >= E) return;
  int lane = threadIdx.x & 31;
  long long r = rows[e], c = cols[e];
  float norm = dinv[r] * ew[e] * dinv[c];
  const float* src = xw + (size_t)r * F;
  float*       dst = out + (size_t)c * F;
  for (int f = lane; f < F; f += 32)
    atomicAdd(&dst[f], norm * src[f]);
}

// ---------------------------------------------------------------------------
// ReLU + f32 -> bf16 with row padding: h[r][0..ldd) from a[r][0..F), pad 0
// ---------------------------------------------------------------------------
__global__ __launch_bounds__(256)
void k_relu_pad_bf(const float* __restrict__ in, uint16_t* __restrict__ out,
                   int rows, int F, int ldd) {
  long long idx = (long long)blockIdx.x * blockDim.x + threadIdx.x;
  if (idx >= (long long)rows * ldd) return;
  int r = (int)(idx / ldd), c = (int)(idx % ldd);
  float v = (c < F) ? fmaxf(in[(size_t)r * F + c], 0.0f) : 0.0f;
  out[idx] = f2bf(v);
}

// ---------------------------------------------------------------------------
// Orchestration
// ---------------------------------------------------------------------------
static inline int ceil_to(int x, int m) { return ((x + m - 1) / m) * m; }

extern "C" void kernel_launch(void* const* d_in, const int* in_sizes, int n_in,
                              void* d_out, int out_size, void* d_ws, size_t ws_size,
                              hipStream_t stream)
{
  const float*     doc   = (const float*)d_in[0];
  const float*     wordf = (const float*)d_in[1];
  const long long* ei    = (const long long*)d_in[2];   // int64 per reference
  const float*     ew    = (const float*)d_in[3];
  const float*     Wlin  = (const float*)d_in[4];
  const float*     blin  = (const float*)d_in[5];
  const float*     W1    = (const float*)d_in[6];
  const float*     b1    = (const float*)d_in[7];
  const float*     W2    = (const float*)d_in[8];
  const float*     b2    = (const float*)d_in[9];
  float*           out   = (float*)d_out;

  const int D_IN = 768, D_W = 300, D_H1 = 200, D_H2 = 100;
  const int KP_W  = ceil_to(D_W, 32);    // 320
  const int KP_IN = D_IN;                // 768
  const int KP_H1 = ceil_to(D_H1, 32);   // 224
  const int NP_IN = ceil_to(D_IN, 64);   // 768
  const int NP_H1 = ceil_to(D_H1, 64);   // 256
  const int NP_H2 = ceil_to(D_H2, 64);   // 128

  const int nDoc  = in_sizes[0] / D_IN;
  const int nWord = in_sizes[1] / D_W;
  const long long E = (long long)in_sizes[3];
  const int N = nDoc + nWord;
  const long long* rows = ei;
  const long long* cols = ei + E;

  // ---- workspace carve-out (with phase-based reuse) ----
  char* ws = (char*)d_ws;
  size_t off = 0;
  auto carve = [&](size_t bytes) -> void* {
    void* p = ws + off;
    off += (bytes + 255) & ~(size_t)255;
    return p;
  };
  float*    deg    = (float*)   carve((size_t)N * 4);
  float*    dinv   = (float*)   carve((size_t)N * 4);
  uint16_t* wlin_t = (uint16_t*)carve((size_t)NP_IN * KP_W * 2);
  uint16_t* w1_t   = (uint16_t*)carve((size_t)NP_H1 * KP_IN * 2);
  uint16_t* w2_t   = (uint16_t*)carve((size_t)NP_H2 * KP_H1 * 2);
  uint16_t* x_bf   = (uint16_t*)carve((size_t)N * D_IN * 2);   // reused as h1_bf
  size_t wfB  = (size_t)nWord * KP_W * 2;
  size_t aggB = (size_t)N * D_H1 * 4;
  void*     bufA = carve(wfB > aggB ? wfB : aggB);             // wf_bf then agg1
  float*    xw   = (float*)  carve((size_t)N * D_H1 * 4);      // xw1 then xw2
  uint16_t* wf_bf = (uint16_t*)bufA;
  float*    agg1  = (float*)bufA;
  uint16_t* h1_bf = x_bf;

  // ---- degree / normalization ----
  hipMemsetAsync(deg, 0, (size_t)N * 4, stream);

  long long n;
  n = (long long)nWord * KP_W;
  k_pad_bf<<<(unsigned)((n + 255) / 256), 256, 0, stream>>>(wordf, wf_bf, nWord, D_W, KP_W);
  n = (long long)nDoc * D_IN;
  k_pad_bf<<<(unsigned)((n + 255) / 256), 256, 0, stream>>>(doc, x_bf, nDoc, D_IN, D_IN);

  n = (long long)NP_IN * KP_W;
  k_wt_bf<<<(unsigned)((n + 255) / 256), 256, 0, stream>>>(Wlin, wlin_t, D_W, D_IN, KP_W, NP_IN);
  n = (long long)NP_H1 * KP_IN;
  k_wt_bf<<<(unsigned)((n + 255) / 256), 256, 0, stream>>>(W1, w1_t, D_IN, D_H1, KP_IN, NP_H1);
  n = (long long)NP_H2 * KP_H1;
  k_wt_bf<<<(unsigned)((n + 255) / 256), 256, 0, stream>>>(W2, w2_t, D_H1, D_H2, KP_H1, NP_H2);

  k_deg<<<(unsigned)((E + 255) / 256), 256, 0, stream>>>(cols, ew, deg, E);
  k_dinv<<<(unsigned)((N + 255) / 256), 256, 0, stream>>>(deg, dinv, N);

  // ---- GEMM1: wf = word_features @ W_lin + b_lin -> bf16 rows of x ----
  {
    dim3 g(D_IN / BN, (nWord + BM - 1) / BM);
    k_gemm_bf16_wmma<<<g, 256, 0, stream>>>(wf_bf, wlin_t, blin,
                                            nullptr, x_bf + (size_t)nDoc * D_IN,
                                            D_IN, nWord, D_IN, KP_W);
  }
  // ---- GEMM2: xw1 = x @ W1 (bias deferred to aggregation) ----
  {
    dim3 g(NP_H1 / BN, (N + BM - 1) / BM);
    k_gemm_bf16_wmma<<<g, 256, 0, stream>>>(x_bf, w1_t, nullptr,
                                            xw, nullptr,
                                            D_H1, N, D_H1, KP_IN);
  }
  // ---- conv1 aggregation ----
  long long nf1 = (long long)N * D_H1;
  k_selfinit<<<(unsigned)((nf1 + 255) / 256), 256, 0, stream>>>(xw, dinv, b1, agg1, N, D_H1);
  k_scatter_edges<<<(unsigned)((E + 7) / 8), 256, 0, stream>>>(rows, cols, ew, dinv,
                                                               xw, agg1, E, D_H1);
  // ---- ReLU -> bf16 h1 (reuses x_bf storage, K padded to 224) ----
  n = (long long)N * KP_H1;
  k_relu_pad_bf<<<(unsigned)((n + 255) / 256), 256, 0, stream>>>(agg1, h1_bf, N, D_H1, KP_H1);

  // ---- GEMM3: xw2 = h1 @ W2 (reuses xw storage) ----
  {
    dim3 g(NP_H2 / BN, (N + BM - 1) / BM);
    k_gemm_bf16_wmma<<<g, 256, 0, stream>>>(h1_bf, w2_t, nullptr,
                                            xw, nullptr,
                                            D_H2, N, D_H2, KP_H1);
  }
  // ---- conv2 aggregation directly into d_out ----
  long long nf2 = (long long)N * D_H2;
  k_selfinit<<<(unsigned)((nf2 + 255) / 256), 256, 0, stream>>>(xw, dinv, b2, out, N, D_H2);
  k_scatter_edges<<<(unsigned)((E + 7) / 8), 256, 0, stream>>>(rows, cols, ew, dinv,
                                                               xw, out, E, D_H2);
}
